// POLATransBlock_317827580594
// MI455X (gfx1250) — matrix operand order repaired and checked by reading, not compiled, not run on hardware
//
#include <hip/hip_runtime.h>
#include <hip/hip_bf16.h>

typedef _Float16 v16h __attribute__((ext_vector_type(16)));
typedef _Float16 v8h  __attribute__((ext_vector_type(8)));
typedef float    v8f  __attribute__((ext_vector_type(8)));
typedef int      v4i  __attribute__((ext_vector_type(4)));

#define DIM      256
#define HEADS    8
#define HDIM     32
#define NTOK     (2 * 96 * 96)       // 18432 tokens
#define NWIN     (2 * 144)           // 288 windows total
#define MLP_HID_ 1024
#define QK_SCALE 0.1767766952966369f // 32^-0.5

// ---- gfx1250 async global->LDS path (guarded; falls back to sync loads) ----
#if defined(__has_builtin)
#if __has_builtin(__builtin_amdgcn_global_load_async_to_lds_b128)
#define HAVE_ASYNC_LDS 1
#endif
#endif

#ifdef HAVE_ASYNC_LDS
typedef __attribute__((address_space(1))) v4i* gptr_v4i;
typedef __attribute__((address_space(3))) v4i* lptr_v4i;
__device__ __forceinline__ void async_copy_b128(void* lds, const void* g) {
  __builtin_amdgcn_global_load_async_to_lds_b128(
      (gptr_v4i)(uintptr_t)g, (lptr_v4i)(uint32_t)(uintptr_t)lds, 0, 0);
}
__device__ __forceinline__ void async_wait0() {
#if __has_builtin(__builtin_amdgcn_s_wait_asynccnt)
  __builtin_amdgcn_s_wait_asynccnt(0);
#else
  asm volatile("s_wait_asynccnt 0" ::: "memory");
#endif
}
#endif

// ---------------------------------------------------------------------------
// WMMA fragment loaders (wave32, 16x16x32 f16)
// A 16x32: lane M = l&15; elems 0-15 = K{hi*16 .. hi*16+15} contiguous
// B 32x16: lane N = l&15; elems 0-15 = K{hi*16 .. hi*16+15} contiguous
// ---------------------------------------------------------------------------
__device__ __forceinline__ v16h frag_a(const _Float16* __restrict__ A, int ld,
                                       int row0, int k0) {
  int lane = threadIdx.x & 31;
  const _Float16* p = A + (size_t)(row0 + (lane & 15)) * ld + k0 + ((lane >> 4) << 4);
  union { v16h v; v8h h[2]; } u;
  u.h[0] = *(const v8h*)(p);
  u.h[1] = *(const v8h*)(p + 8);
  return u.v;
}

__device__ __forceinline__ v16h frag_b(const _Float16* __restrict__ B, int ld,
                                       int col0, int k0) {
  int lane = threadIdx.x & 31;
  const _Float16* p = B + (size_t)(col0 + (lane & 15)) * ld + k0 + ((lane >> 4) << 4);
  union { v16h v; v8h h[2]; } u;
  u.h[0] = *(const v8h*)(p);
  u.h[1] = *(const v8h*)(p + 8);
  return u.v;
}

#define WMMA_F16(a, b, c) \
  __builtin_amdgcn_wmma_f32_16x16x32_f16(false, (a), false, (b), (short)0, (c), false, false)

// ---------------------------------------------------------------------------
// f32 -> f16 weight conversion
// ---------------------------------------------------------------------------
__global__ void cvt_f16_kernel(const float* __restrict__ src,
                               _Float16* __restrict__ dst, int n) {
  int i = blockIdx.x * blockDim.x + threadIdx.x;
  if (i < n) dst[i] = (_Float16)src[i];
}

// ---------------------------------------------------------------------------
// LayerNorm over C=256: one wave per token, 8 elems per lane
// ---------------------------------------------------------------------------
__global__ __launch_bounds__(256) void ln_kernel(const float* __restrict__ x,
    const float* __restrict__ w, const float* __restrict__ b,
    _Float16* __restrict__ out, int ntok) {
  int tok = blockIdx.x * 8 + (threadIdx.x >> 5);
  if (tok >= ntok) return;
  int lane = threadIdx.x & 31;
  const float* xp = x + (size_t)tok * DIM + lane * 8;
  float v[8], s = 0.f, s2 = 0.f;
#pragma unroll
  for (int i = 0; i < 8; ++i) { v[i] = xp[i]; s += v[i]; s2 += v[i] * v[i]; }
#pragma unroll
  for (int m = 16; m >= 1; m >>= 1) {
    s  += __shfl_xor(s,  m, 32);
    s2 += __shfl_xor(s2, m, 32);
  }
  float mu  = s * (1.f / DIM);
  float var = s2 * (1.f / DIM) - mu * mu;
  float rs  = rsqrtf(var + 1e-5f);
  _Float16* op = out + (size_t)tok * DIM + lane * 8;
#pragma unroll
  for (int i = 0; i < 8; ++i) {
    int c = lane * 8 + i;
    op[i] = (_Float16)((v[i] - mu) * rs * w[c] + b[c]);
  }
}

// ---------------------------------------------------------------------------
// Generic WMMA GEMM: out[M,N] = A[M,K] @ W[N,K]^T (+ bias)(epilogue)
// Each wave computes a 32x32 tile (2x2 WMMA subtiles) -> 2x operand reuse,
// 4 back-to-back v_wmma per k-step. Prefetch next k-slice into L2/WGP$.
// EPI: 0 = f16 store; 1 = f16 store, *scale; 2 = GELU -> f16; 3 = +res -> f32
// ---------------------------------------------------------------------------
template <int EPI>
__global__ __launch_bounds__(128) void gemm_wmma_kernel(
    const _Float16* __restrict__ A, const _Float16* __restrict__ W,
    const float* __restrict__ bias, void* __restrict__ outv,
    const float* __restrict__ residual, float scale, int M, int N, int K) {
  int wave = threadIdx.x >> 5;
  int lane = threadIdx.x & 31;
  int half = lane >> 4, lc = lane & 15;
  int ntiles = N >> 5;
  int tile = blockIdx.x * 4 + wave;
  if (tile >= (M >> 5) * ntiles) return;
  int mt = tile / ntiles, nt = tile - mt * ntiles;
  int m0 = mt << 5, n0 = nt << 5;

  v8f acc00 = {}, acc01 = {}, acc10 = {}, acc11 = {};
  for (int k0 = 0; k0 < K; k0 += 32) {
    if (k0 + 32 < K) {
      __builtin_prefetch(A + (size_t)(m0 + lc) * K + k0 + 32, 0, 1);
      __builtin_prefetch(W + (size_t)(n0 + lc) * K + k0 + 32, 0, 1);
    }
    v16h a0 = frag_a(A, K, m0, k0);
    v16h a1 = frag_a(A, K, m0 + 16, k0);
    v16h b0 = frag_b(W, K, n0, k0);
    v16h b1 = frag_b(W, K, n0 + 16, k0);
    acc00 = WMMA_F16(a0, b0, acc00);
    acc01 = WMMA_F16(a0, b1, acc01);
    acc10 = WMMA_F16(a1, b0, acc10);
    acc11 = WMMA_F16(a1, b1, acc11);
  }

#pragma unroll
  for (int i = 0; i < 2; ++i) {
#pragma unroll
    for (int j = 0; j < 2; ++j) {
      const v8f& acc = (i == 0) ? (j == 0 ? acc00 : acc01)
                                : (j == 0 ? acc10 : acc11);
      int col   = n0 + j * 16 + lc;
      int rbase = m0 + i * 16 + half * 8;
      float bv = bias[col];
#pragma unroll
      for (int r = 0; r < 8; ++r) {
        int row = rbase + r;
        float v = acc[r] + bv;
        if (EPI == 1) v *= scale;
        if (EPI == 2) v = 0.5f * v * (1.f + erff(v * 0.7071067811865476f));
        if (EPI == 3) {
          ((float*)outv)[(size_t)row * N + col] = v + residual[(size_t)row * N + col];
        } else {
          ((_Float16*)outv)[(size_t)row * N + col] = (_Float16)v;
        }
      }
    }
  }
}

// ---------------------------------------------------------------------------
// Neighborhood window attention, flash-style online softmax.
// Block = (window, head); 128 threads = 4 waves; wave m owns queries m*16..+15.
// 576 keys processed in 18 chunks of 32 (2 key tiles per chunk).
// V chunk staged global->LDS with async b128 copies (when available), then
// transposed LDS->LDS with OOB keys substituted by wv_b (matches the
// reference's zero-padded kv @ W^T + b). K fragments loaded from global.
// ---------------------------------------------------------------------------
__global__ __launch_bounds__(128) void attn_kernel(
    const _Float16* __restrict__ Qb, const _Float16* __restrict__ Kb,
    const _Float16* __restrict__ Vb, const float* __restrict__ kbias,
    const float* __restrict__ vbias, const float* __restrict__ rpb,
    _Float16* __restrict__ out) {
  __shared__ _Float16 Vraw[32][32];    // [key][dim] raw staged chunk (b128 rows)
  __shared__ _Float16 Vt[32][48];      // [dim][key] transposed V chunk
  __shared__ _Float16 P[4][16][40];    // per-wave probability tile 16x32

  int win = blockIdx.x, head = blockIdx.y;
  int bat = win / 144, rem = win - bat * 144;
  int wi = rem / 12, wj = rem - wi * 12;
  int wave = threadIdx.x >> 5, lane = threadIdx.x & 31;
  int half = lane >> 4, lc = lane & 15;

  int q0 = wave * 16;
  int ybase = wi * 8 - 8, xbase = wj * 8 - 8;

  // Q fragment (A operand), constant across the key loop
  v16h qfrag;
  {
    int qa = q0 + lc;
    int qtok = bat * 9216 + (wi * 8 + (qa >> 3)) * 96 + (wj * 8 + (qa & 7));
    const _Float16* p = Qb + (size_t)qtok * DIM + head * HDIM + half * 16;
    union { v16h v; v8h h[2]; } u;
    u.h[0] = *(const v8h*)(p);
    u.h[1] = *(const v8h*)(p + 8);
    qfrag = u.v;
  }

  v8f o0 = {}, o1 = {};
  float mrun[8], lrun[8];
#pragma unroll
  for (int r = 0; r < 8; ++r) { mrun[r] = -1e30f; lrun[r] = 0.f; }

  for (int kc = 0; kc < 18; ++kc) {
    int kbase = kc * 32;
    __syncthreads();
    // ---- stage raw V chunk (32 keys x 32 dims, key-major) ----
#ifdef HAVE_ASYNC_LDS
    {
      int kk  = threadIdx.x >> 2;       // 0..31
      int seg = threadIdx.x & 3;        // 16B segment within 64B row
      int key = kbase + kk;
      int ky = key / 24, kx = key - ky * 24;
      int y = ybase + ky, x = xbase + kx;
      int tok = ((unsigned)y < 96u && (unsigned)x < 96u)
                    ? (bat * 9216 + y * 96 + x) : 0;  // clamp; fixed in transpose
      async_copy_b128(&Vraw[kk][seg * 8],
                      Vb + (size_t)tok * DIM + head * HDIM + seg * 8);
      async_wait0();
    }
#else
    for (int i = threadIdx.x; i < 32 * 32; i += 128) {
      int kk = i >> 5, d = i & 31;
      int key = kbase + kk;
      int ky = key / 24, kx = key - ky * 24;
      int y = ybase + ky, x = xbase + kx;
      int tok = ((unsigned)y < 96u && (unsigned)x < 96u)
                    ? (bat * 9216 + y * 96 + x) : 0;
      Vraw[kk][d] = Vb[(size_t)tok * DIM + head * HDIM + d];
    }
#endif
    __syncthreads();
    // ---- transpose into Vt with OOB bias substitution ----
    for (int i = threadIdx.x; i < 32 * 32; i += 128) {
      int kk = i >> 5, d = i & 31;
      int key = kbase + kk;
      int ky = key / 24, kx = key - ky * 24;
      int y = ybase + ky, x = xbase + kx;
      _Float16 val;
      if ((unsigned)y < 96u && (unsigned)x < 96u) val = Vraw[kk][d];
      else                                        val = (_Float16)vbias[head * HDIM + d];
      Vt[d][kk] = val;
    }

    // ---- scores: two 16-key tiles via WMMA (K fragments from global) ----
    v8f s0 = {}, s1 = {};
#pragma unroll
    for (int t = 0; t < 2; ++t) {
      int key = kbase + t * 16 + lc;
      int ky = key / 24, kx = key - ky * 24;
      int y = ybase + ky, x = xbase + kx;
      union { v16h v; v8h h[2]; } u;
      if ((unsigned)y < 96u && (unsigned)x < 96u) {
        int tok = bat * 9216 + y * 96 + x;
        const _Float16* p = Kb + (size_t)tok * DIM + head * HDIM + half * 16;
        u.h[0] = *(const v8h*)(p);
        u.h[1] = *(const v8h*)(p + 8);
      } else {
        const float* p = kbias + head * HDIM + half * 16;
#pragma unroll
        for (int i = 0; i < 16; ++i) u.v[i] = (_Float16)p[i];
      }
      if (t == 0) s0 = WMMA_F16(qfrag, u.v, s0);
      else        s1 = WMMA_F16(qfrag, u.v, s1);
    }

    // ---- relative position bias: idx = (dy+23)*31 + (dx+23) ----
    {
      int k0i = kbase + lc, k1i = kbase + 16 + lc;
      int ky0 = k0i / 24, kx0 = k0i - ky0 * 24;
      int ky1 = k1i / 24, kx1 = k1i - ky1 * 24;
#pragma unroll
      for (int r = 0; r < 8; ++r) {
        int q = q0 + r + half * 8;
        int qy = (q >> 3) & 7, qx = q & 7;
        s0[r] = s0[r] + rpb[((qy - ky0 + 23) * 31 + (qx - kx0 + 23)) * HEADS + head];
        s1[r] = s1[r] + rpb[((qy - ky1 + 23) * 31 + (qx - kx1 + 23)) * HEADS + head];
      }
    }

    // ---- online softmax update (row = 16 lanes within half-wave) ----
    float p0[8], p1[8];
#pragma unroll
    for (int r = 0; r < 8; ++r) {
      float mx = fmaxf(s0[r], s1[r]);
#pragma unroll
      for (int m = 8; m >= 1; m >>= 1) mx = fmaxf(mx, __shfl_xor(mx, m, 32));
      float mnew = fmaxf(mrun[r], mx);
      float sc = __expf(mrun[r] - mnew);
      float e0 = __expf(s0[r] - mnew);
      float e1 = __expf(s1[r] - mnew);
      float rs = e0 + e1;
#pragma unroll
      for (int m = 8; m >= 1; m >>= 1) rs += __shfl_xor(rs, m, 32);
      lrun[r] = lrun[r] * sc + rs;
      mrun[r] = mnew;
      o0[r] = o0[r] * sc;
      o1[r] = o1[r] * sc;
      p0[r] = e0;
      p1[r] = e1;
    }

    // ---- D-layout -> A-layout transpose of P through LDS ----
#pragma unroll
    for (int r = 0; r < 8; ++r) {
      int row = r + half * 8;
      P[wave][row][lc]      = (_Float16)p0[r];
      P[wave][row][16 + lc] = (_Float16)p1[r];
    }
    __syncthreads();

    v16h pa;
    {
      const _Float16* pp = &P[wave][lc][0];
      union { v16h v; v8h h[2]; } u;
      u.h[0] = *(const v8h*)(pp + half * 16);
      u.h[1] = *(const v8h*)(pp + half * 16 + 8);
      pa = u.v;
    }
    // ---- O += P @ V (two 16-dim N tiles, B operand from transposed LDS) ----
#pragma unroll
    for (int t = 0; t < 2; ++t) {
      const _Float16* vp = &Vt[t * 16 + lc][0] + half * 16;
      union { v16h v; v8h h[2]; } u;
      u.h[0] = *(const v8h*)(vp);
      u.h[1] = *(const v8h*)(vp + 8);
      if (t == 0) o0 = WMMA_F16(pa, u.v, o0);
      else        o1 = WMMA_F16(pa, u.v, o1);
    }
  }

  // ---- normalize & store attention output (token-major, head slice) ----
#pragma unroll
  for (int r = 0; r < 8; ++r) {
    int q = q0 + r + half * 8;
    int tok = bat * 9216 + (wi * 8 + (q >> 3)) * 96 + (wj * 8 + (q & 7));
    float inv = 1.f / lrun[r];
    out[(size_t)tok * DIM + head * HDIM + lc]      = (_Float16)(o0[r] * inv);
    out[(size_t)tok * DIM + head * HDIM + 16 + lc] = (_Float16)(o1[r] * inv);
  }
}

// ---------------------------------------------------------------------------
// Host launcher
// ---------------------------------------------------------------------------
extern "C" void kernel_launch(void* const* d_in, const int* in_sizes, int n_in,
                              void* d_out, int out_size, void* d_ws, size_t ws_size,
                              hipStream_t stream) {
  const float* x    = (const float*)d_in[0];
  const float* n1w  = (const float*)d_in[3];
  const float* n1b  = (const float*)d_in[4];
  const float* wq_w = (const float*)d_in[5];
  const float* wq_b = (const float*)d_in[6];
  const float* wk_w = (const float*)d_in[7];
  const float* wk_b = (const float*)d_in[8];
  const float* wv_w = (const float*)d_in[9];
  const float* wv_b = (const float*)d_in[10];
  const float* pj_w = (const float*)d_in[11];
  const float* pj_b = (const float*)d_in[12];
  const float* rpb  = (const float*)d_in[13];
  const float* n2w  = (const float*)d_in[14];
  const float* n2b  = (const float*)d_in[15];
  const float* f1w  = (const float*)d_in[16];
  const float* f1b  = (const float*)d_in[17];
  const float* f2w  = (const float*)d_in[18];
  const float* f2b  = (const float*)d_in[19];

  char* ws = (char*)d_ws;
  size_t off = 0;
  auto alloc = [&](size_t bytes) -> char* {
    char* p = ws + off;
    off += (bytes + 255) & ~(size_t)255;
    return p;
  };
  _Float16* wqh  = (_Float16*)alloc((size_t)DIM * DIM * 2);
  _Float16* wkh  = (_Float16*)alloc((size_t)DIM * DIM * 2);
  _Float16* wvh  = (_Float16*)alloc((size_t)DIM * DIM * 2);
  _Float16* pjh  = (_Float16*)alloc((size_t)DIM * DIM * 2);
  _Float16* f1h  = (_Float16*)alloc((size_t)MLP_HID_ * DIM * 2);
  _Float16* f2h  = (_Float16*)alloc((size_t)DIM * MLP_HID_ * 2);
  _Float16* hA   = (_Float16*)alloc((size_t)NTOK * DIM * 2);
  _Float16* Qb   = (_Float16*)alloc((size_t)NTOK * DIM * 2);
  _Float16* Kb   = (_Float16*)alloc((size_t)NTOK * DIM * 2);
  _Float16* Vb   = (_Float16*)alloc((size_t)NTOK * DIM * 2);
  _Float16* attn = (_Float16*)alloc((size_t)NTOK * DIM * 2);
  float*    x1   = (float*)alloc((size_t)NTOK * DIM * 4);
  _Float16* h2   = (_Float16*)alloc((size_t)NTOK * DIM * 2);
  _Float16* hid  = (_Float16*)alloc((size_t)NTOK * MLP_HID_ * 2);
  (void)ws_size; (void)in_sizes; (void)n_in; (void)out_size;

  // weight conversion
  cvt_f16_kernel<<<(DIM * DIM + 255) / 256, 256, 0, stream>>>(wq_w, wqh, DIM * DIM);
  cvt_f16_kernel<<<(DIM * DIM + 255) / 256, 256, 0, stream>>>(wk_w, wkh, DIM * DIM);
  cvt_f16_kernel<<<(DIM * DIM + 255) / 256, 256, 0, stream>>>(wv_w, wvh, DIM * DIM);
  cvt_f16_kernel<<<(DIM * DIM + 255) / 256, 256, 0, stream>>>(pj_w, pjh, DIM * DIM);
  cvt_f16_kernel<<<(MLP_HID_ * DIM + 255) / 256, 256, 0, stream>>>(f1w, f1h, MLP_HID_ * DIM);
  cvt_f16_kernel<<<(MLP_HID_ * DIM + 255) / 256, 256, 0, stream>>>(f2w, f2h, MLP_HID_ * DIM);

  // LN1
  ln_kernel<<<NTOK / 8, 256, 0, stream>>>(x, n1w, n1b, hA, NTOK);

  // QKV projections (32x32 tiles per wave)
  int tiles_d = (NTOK / 32) * (DIM / 32);
  gemm_wmma_kernel<1><<<(tiles_d + 3) / 4, 128, 0, stream>>>(
      hA, wqh, wq_b, Qb, nullptr, QK_SCALE, NTOK, DIM, DIM);
  gemm_wmma_kernel<0><<<(tiles_d + 3) / 4, 128, 0, stream>>>(
      hA, wkh, wk_b, Kb, nullptr, 1.f, NTOK, DIM, DIM);
  gemm_wmma_kernel<0><<<(tiles_d + 3) / 4, 128, 0, stream>>>(
      hA, wvh, wv_b, Vb, nullptr, 1.f, NTOK, DIM, DIM);

  // windowed neighborhood attention
  attn_kernel<<<dim3(NWIN, HEADS), 128, 0, stream>>>(Qb, Kb, Vb, wk_b, wv_b, rpb, attn);

  // output projection + residual (f32)
  gemm_wmma_kernel<3><<<(tiles_d + 3) / 4, 128, 0, stream>>>(
      attn, pjh, pj_b, x1, x, 1.f, NTOK, DIM, DIM);

  // LN2
  ln_kernel<<<NTOK / 8, 256, 0, stream>>>(x1, n2w, n2b, h2, NTOK);

  // MLP
  int tiles_h = (NTOK / 32) * (MLP_HID_ / 32);
  gemm_wmma_kernel<2><<<(tiles_h + 3) / 4, 128, 0, stream>>>(
      h2, f1h, f1b, hid, nullptr, 1.f, NTOK, MLP_HID_, DIM);
  gemm_wmma_kernel<3><<<(tiles_d + 3) / 4, 128, 0, stream>>>(
      hid, f2h, f2b, d_out, x1, 1.f, NTOK, DIM, MLP_HID_);
}
